// CausalSelfAttention_4157528342540
// MI455X (gfx1250) — matrix operand order repaired and checked by reading, not compiled, not run on hardware
//
#include <hip/hip_runtime.h>

// ---------------------------------------------------------------------------
// Causal self-attention for MI455X (gfx1250, wave32, WMMA).
// B=4, S=2048, E=1024, H=16, D=64.
// Pipeline: f32->bf16 convert, QKV GEMM (bf16 WMMA, f32 accum, double-buffered
// LDS, 64x64 wave tiles), flash attention (bf16 WMMA incl. WMMA row-sums),
// out-proj GEMM -> f32.
// ---------------------------------------------------------------------------

typedef __attribute__((ext_vector_type(16))) __bf16 v16bf;
typedef __attribute__((ext_vector_type(8)))  __bf16 v8bf;
typedef __attribute__((ext_vector_type(8)))  float  v8f;

union V16U { v16bf v; v8bf h[2]; };
union PK2  { __bf16 h[2]; unsigned u; };
union FI   { float f; int i; };

__device__ __forceinline__ v8f wmma_bf16(v16bf a, v16bf b, v8f c) {
  // D(f32,16x16) = A(bf16,16x32) * B(bf16,32x16) + C
  return __builtin_amdgcn_wmma_f32_16x16x32_bf16(false, a, false, b,
                                                 (short)0, c, false, false);
}

// A-fragment (16x32 bf16, row-major source with given element stride).
// Lane row M=lane&15: elems 0..15 = K[hf*16..hf*16+15] contiguous.
__device__ __forceinline__ v16bf load_a_frag(const __bf16* base, int stride, int lane) {
  const int hf = lane >> 4, r = lane & 15;
  V16U u;
  u.h[0] = *(const v8bf*)(base + r * stride + hf * 16);
  u.h[1] = *(const v8bf*)(base + r * stride + hf * 16 + 8);
  return u.v;
}

// B-fragment (32x16 bf16) from a K-transposed tile: baseT[n][k]; lane n holds
// column N=n with K-run [hf*16 .. hf*16+15] contiguous.
__device__ __forceinline__ v16bf load_b_frag(const __bf16* baseT, int stride, int lane) {
  const int hf = lane >> 4, n = lane & 15;
  return *(const v16bf*)(baseT + n * stride + hf * 16);
}

// Butterfly max over 16-lane halves using LDS swizzle (group-of-32 xor mode:
// offset = {or=0, xor_mask<<5, and_mask=0x1f}).
__device__ __forceinline__ float rowmax16(float v) {
  FI a, r;
  a.f = v; r.i = __builtin_amdgcn_ds_swizzle(a.i, 0x003F); v = fmaxf(v, r.f);
  a.f = v; r.i = __builtin_amdgcn_ds_swizzle(a.i, 0x005F); v = fmaxf(v, r.f);
  a.f = v; r.i = __builtin_amdgcn_ds_swizzle(a.i, 0x009F); v = fmaxf(v, r.f);
  a.f = v; r.i = __builtin_amdgcn_ds_swizzle(a.i, 0x011F); v = fmaxf(v, r.f);
  return v;
}

// ---------------------------------------------------------------------------
// f32 -> bf16 conversion (grid-stride)
// ---------------------------------------------------------------------------
__global__ void cvt_f32_to_bf16(const float* __restrict__ src,
                                __bf16* __restrict__ dst, int n) {
  int i = blockIdx.x * blockDim.x + threadIdx.x;
  int stride = gridDim.x * blockDim.x;
  for (; i < n; i += stride) dst[i] = (__bf16)src[i];
}

// ---------------------------------------------------------------------------
// bf16 GEMM: C[M,N] = A[M,K] * B[K,N] + bias[N]
// Block tile 256x128, k-step 32, 256 threads (8 waves), wave tile 64x64
// (16 WMMA per k-step per wave). Double-buffered LDS pipeline: 1 barrier per
// k-step, global loads issued before the WMMA phase, prefetch 2 tiles ahead.
// ---------------------------------------------------------------------------
template <typename OutT>
__global__ __launch_bounds__(256)
void gemm_bf16_kernel(const __bf16* __restrict__ A, const __bf16* __restrict__ B,
                      const float* __restrict__ bias, OutT* __restrict__ C,
                      int M, int N, int K) {
  __shared__ __bf16 As[2][256][32];   // row-major [m][k]   (32 KB)
  __shared__ __bf16 Bt[2][128][32];   // transposed [n][k]  (16 KB)

  const int tid  = threadIdx.x;
  const int lane = tid & 31;
  const int w    = tid >> 5;
  const int wm   = w >> 1;            // 0..3 -> 64-row slices
  const int wn   = w & 1;             // 0..1 -> 64-col slices
  const int m0   = blockIdx.y * 256;
  const int n0   = blockIdx.x * 128;

  // staging thread mappings
  const int arow = tid;                             // A: 1 thr/row, 32 bf16
  const int kp   = (tid >> 4) * 2;                  // B: even k of a k-pair
  const int n8   = (tid & 15) * 8;                  // B: 8 columns

  v8f acc[4][4];
#pragma unroll
  for (int mi = 0; mi < 4; ++mi)
#pragma unroll
    for (int ni = 0; ni < 4; ++ni) acc[mi][ni] = {};

  v16bf areg0, areg1;
  v8bf  breg0, breg1;

  auto gload = [&](int k0) {
    const __bf16* ap = &A[(size_t)(m0 + arow) * K + k0];
    areg0 = *(const v16bf*)(ap);
    areg1 = *(const v16bf*)(ap + 16);
    breg0 = *(const v8bf*)&B[(size_t)(k0 + kp) * N + n0 + n8];
    breg1 = *(const v8bf*)&B[(size_t)(k0 + kp + 1) * N + n0 + n8];
  };
  auto lstore = [&](int buf) {
    *(v16bf*)&As[buf][arow][0]  = areg0;
    *(v16bf*)&As[buf][arow][16] = areg1;
#pragma unroll
    for (int j = 0; j < 8; ++j) {           // pack k-pair -> one b32 store
      PK2 p; p.h[0] = breg0[j]; p.h[1] = breg1[j];
      *(unsigned*)&Bt[buf][n8 + j][kp] = p.u;
    }
  };

  gload(0);
  lstore(0);
  __syncthreads();

  int buf = 0;
  for (int k0 = 0; k0 < K; k0 += 32) {
    const bool has_next = (k0 + 32 < K);
    if (has_next) gload(k0 + 32);
    if (k0 + 64 < K) {
      __builtin_prefetch(&A[(size_t)(m0 + arow) * K + k0 + 64], 0, 0);
      __builtin_prefetch(&B[(size_t)(k0 + 64 + kp) * N + n0 + n8], 0, 0);
    }

    v16bf af[4];
#pragma unroll
    for (int mi = 0; mi < 4; ++mi)
      af[mi] = load_a_frag(&As[buf][wm * 64 + mi * 16][0], 32, lane);
#pragma unroll
    for (int ni = 0; ni < 4; ++ni) {
      const v16bf bfr = load_b_frag(&Bt[buf][wn * 64 + ni * 16][0], 32, lane);
#pragma unroll
      for (int mi = 0; mi < 4; ++mi)
        acc[mi][ni] = wmma_bf16(af[mi], bfr, acc[mi][ni]);
    }

    if (has_next) lstore(buf ^ 1);
    __syncthreads();
    buf ^= 1;
  }

  // Epilogue: C/D layout -> row = hf*8 + i, col = lane&15 per 16x16 tile.
  const int hf = lane >> 4, nl = lane & 15;
#pragma unroll
  for (int mi = 0; mi < 4; ++mi) {
#pragma unroll
    for (int ni = 0; ni < 4; ++ni) {
      const int row = m0 + wm * 64 + mi * 16 + hf * 8;
      const int col = n0 + wn * 64 + ni * 16 + nl;
      const float bcol = bias[col];
#pragma unroll
      for (int i = 0; i < 8; ++i) {
        float v = acc[mi][ni][i] + bcol;
        C[(size_t)(row + i) * N + col] = (OutT)v;
      }
    }
  }
}

// ---------------------------------------------------------------------------
// Flash attention: one block = 64 query rows of one (b,h). 128 threads = 4
// waves; wave w owns query rows [q0+16w, q0+16w+16).
// Softmax in base-2 (scale = log2(e)/sqrt(D) folded into logits); row sums
// computed with a WMMA against an all-ones B fragment.
// ---------------------------------------------------------------------------
__global__ __launch_bounds__(128)
void attn_kernel(const __bf16* __restrict__ qkv, __bf16* __restrict__ y) {
  constexpr int S = 2048, E = 1024, D = 64;
  constexpr float SCALE = 0.18033688011112042f;   // log2(e) / sqrt(64)
  __shared__ __bf16 Qs[64][64];        // [q][d]
  __shared__ __bf16 Ks[32][64];        // [key][d]  (B-frag layout for Q*K^T)
  __shared__ __bf16 Vt[64][32];        // [d][key]  (B-frag layout for P*V)
  __shared__ __bf16 Ps[4][16][32];     // per-wave P tile [q][key]

  const int tid  = threadIdx.x;
  const int lane = tid & 31;
  const int w    = tid >> 5;
  const int bh   = blockIdx.x;
  const int b    = bh >> 4;
  const int h    = bh & 15;
  const int q0   = blockIdx.y * 64;
  const int hf   = lane >> 4, nl = lane & 15;

  const size_t baseQ = (size_t)b * S * (3 * E) + (size_t)h * D;

  // All-ones B fragment for WMMA row sums.
  v16bf ones;
#pragma unroll
  for (int e = 0; e < 16; ++e) ones[e] = (__bf16)1.0f;

  // Load 64x64 Q tile.
  {
    const int row = tid >> 1, seg = tid & 1;
    const size_t g = baseQ + (size_t)(q0 + row) * (3 * E) + seg * 32;
    *(v16bf*)&Qs[row][seg * 32]      = *(const v16bf*)&qkv[g];
    *(v16bf*)&Qs[row][seg * 32 + 16] = *(const v16bf*)&qkv[g + 16];
  }
  __syncthreads();

  const v16bf aq0 = load_a_frag(&Qs[w * 16][0], 64, lane);   // d 0..31
  const v16bf aq1 = load_a_frag(&Qs[w * 16][32], 64, lane);  // d 32..63

  float m_st[8], l_st[8], sc_sv[8];
  v8f acc[4];
#pragma unroll
  for (int i = 0; i < 8; ++i) { m_st[i] = -3.0e38f; l_st[i] = 0.0f; }
#pragma unroll
  for (int j = 0; j < 4; ++j) acc[j] = {};

  // staging mappings for K (vector rows) and V (transposed, k-pair packed)
  const int kKey = tid >> 2, kSeg = tid & 3;        // K: 4 thr/key, 16 d each
  const int vKey = (tid >> 3) * 2;                  // V: even key of pair
  const int vD8  = (tid & 7) * 8;                   // V: 8 d values

  const int nkb = (q0 + 64) / 32;   // causal: keys up to end of query block
  for (int kb = 0; kb < nkb; ++kb) {
    const int kk0 = kb * 32;
    __syncthreads();   // previous iteration done reading Ks/Vt

    // Stage K block [key][d].
    {
      const size_t g = baseQ + E + (size_t)(kk0 + kKey) * (3 * E) + kSeg * 16;
      *(v16bf*)&Ks[kKey][kSeg * 16] = *(const v16bf*)&qkv[g];
    }
    // Stage V block transposed [d][key], packing key-pairs into b32 stores.
    {
      const size_t g = baseQ + 2 * E + (size_t)(kk0 + vKey) * (3 * E) + vD8;
      v8bf v0 = *(const v8bf*)&qkv[g];
      v8bf v1 = *(const v8bf*)&qkv[g + 3 * E];
#pragma unroll
      for (int j = 0; j < 8; ++j) {
        PK2 p; p.h[0] = v0[j]; p.h[1] = v1[j];
        *(unsigned*)&Vt[vD8 + j][vKey] = p.u;
      }
    }
    __syncthreads();

    // Scores: two 16x16 key tiles, K-depth 64 = two WMMAs each.
    v8f s0 = {}, s1 = {};
    s0 = wmma_bf16(aq0, load_b_frag(&Ks[0][0], 64, lane), s0);
    s0 = wmma_bf16(aq1, load_b_frag(&Ks[0][32], 64, lane), s0);
    s1 = wmma_bf16(aq0, load_b_frag(&Ks[16][0], 64, lane), s1);
    s1 = wmma_bf16(aq1, load_b_frag(&Ks[16][32], 64, lane), s1);

    // Scale into base-2 domain and apply causal mask (finite -inf surrogate).
    const int col0 = kk0 + nl, col1 = kk0 + 16 + nl;
    const int rowg = q0 + w * 16 + hf * 8;
#pragma unroll
    for (int i = 0; i < 8; ++i) {
      float a = s0[i] * SCALE;
      float c = s1[i] * SCALE;
      if (col0 > rowg + i) a = -3.0e38f;
      if (col1 > rowg + i) c = -3.0e38f;
      s0[i] = a; s1[i] = c;
    }

    // Online softmax: row max via LDS swizzle butterflies, probabilities via
    // v_exp; row sums come from a WMMA below.
#pragma unroll
    for (int i = 0; i < 8; ++i) {
      const float mx = rowmax16(fmaxf(s0[i], s1[i]));
      const float mn = fmaxf(m_st[i], mx);
      const float sc = __builtin_amdgcn_exp2f(m_st[i] - mn);
      const float p0 = __builtin_amdgcn_exp2f(s0[i] - mn);
      const float p1 = __builtin_amdgcn_exp2f(s1[i] - mn);
      m_st[i] = mn;
      sc_sv[i] = sc;
#pragma unroll
      for (int j = 0; j < 4; ++j) acc[j][i] *= sc;
      Ps[w][hf * 8 + i][nl]      = (__bf16)p0;
      Ps[w][hf * 8 + i][16 + nl] = (__bf16)p1;
    }
    __syncthreads();   // Ps visible (C->A layout round-trip complete)

    const v16bf ap = load_a_frag(&Ps[w][0][0], 32, lane);

    // Row sums: D = P @ ones; lane slot [i] holds sum of row hf*8+i.
    v8f rsum = {};
    rsum = wmma_bf16(ap, ones, rsum);
#pragma unroll
    for (int i = 0; i < 8; ++i) l_st[i] = l_st[i] * sc_sv[i] + rsum[i];

    // O += P @ V : A = P (16q x 32k), B = V (32k x 16d) per d-tile.
#pragma unroll
    for (int j = 0; j < 4; ++j)
      acc[j] = wmma_bf16(ap, load_b_frag(&Vt[j * 16][0], 32, lane), acc[j]);
  }

  // Normalize and write y[b, s, h*D + d] as bf16.
#pragma unroll
  for (int i = 0; i < 8; ++i) {
    const float inv = 1.0f / l_st[i];
    const size_t row = (size_t)b * S + q0 + w * 16 + hf * 8 + i;
#pragma unroll
    for (int j = 0; j < 4; ++j)
      y[row * E + h * D + j * 16 + nl] = (__bf16)(acc[j][i] * inv);
  }
}

// ---------------------------------------------------------------------------
// Launch
// ---------------------------------------------------------------------------
extern "C" void kernel_launch(void* const* d_in, const int* in_sizes, int n_in,
                              void* d_out, int out_size, void* d_ws, size_t ws_size,
                              hipStream_t stream) {
  constexpr int B = 4, S = 2048, E = 1024;
  constexpr int M = B * S;          // 8192
  constexpr int N_QKV = 3 * E;      // 3072

  const float* x      = (const float*)d_in[0];
  const float* w_qkv  = (const float*)d_in[1];
  const float* b_qkv  = (const float*)d_in[2];
  const float* w_proj = (const float*)d_in[3];
  const float* b_proj = (const float*)d_in[4];
  float* out = (float*)d_out;

  char* ws = (char*)d_ws;
  __bf16* x_bf     = (__bf16*)(ws);                                  // 16 MB
  __bf16* wqkv_bf  = (__bf16*)(ws + (size_t)16 * 1024 * 1024);       //  6 MB
  __bf16* wproj_bf = (__bf16*)(ws + (size_t)22 * 1024 * 1024);       //  2 MB
  __bf16* qkv_bf   = (__bf16*)(ws + (size_t)24 * 1024 * 1024);       // 48 MB
  __bf16* y_bf     = (__bf16*)(ws + (size_t)72 * 1024 * 1024);       // 16 MB

  // 1) Convert inputs to bf16.
  cvt_f32_to_bf16<<<4096, 256, 0, stream>>>(x, x_bf, M * E);
  cvt_f32_to_bf16<<<2048, 256, 0, stream>>>(w_qkv, wqkv_bf, E * N_QKV);
  cvt_f32_to_bf16<<<1024, 256, 0, stream>>>(w_proj, wproj_bf, E * E);

  // 2) QKV projection: [8192,1024] x [1024,3072] -> bf16 qkv.
  {
    dim3 grid(N_QKV / 128, M / 256);
    gemm_bf16_kernel<__bf16><<<grid, 256, 0, stream>>>(
        x_bf, wqkv_bf, b_qkv, qkv_bf, M, N_QKV, E);
  }

  // 3) Causal flash attention -> bf16 y.
  {
    dim3 grid(B * 16 /*H*/, S / 64);
    attn_kernel<<<grid, 128, 0, stream>>>(qkv_bf, y_bf);
  }

  // 4) Output projection: [8192,1024] x [1024,1024] -> f32 out.
  {
    dim3 grid(E / 128, M / 256);
    gemm_bf16_kernel<float><<<grid, 256, 0, stream>>>(
        y_bf, wproj_bf, b_proj, out, M, E, E);
  }
}
